// S4DBlock_12266426597685
// MI455X (gfx1250) — matrix operand order repaired and checked by reading, not compiled, not run on hardware
//
#include <hip/hip_runtime.h>
#include <hip/hip_bf16.h>

#define BB 4
#define CC 512
#define TT 4096
#define HH 2048
#define N2 32
#define CHUNK 128

typedef __attribute__((ext_vector_type(16))) __bf16 v16bf;
typedef __attribute__((ext_vector_type(8)))  float  v8f;

union Frag { v16bf v; unsigned int u[8]; };

__device__ __forceinline__ unsigned short f32_to_bf16(float f) {
    unsigned int u = __float_as_uint(f);
    unsigned int r = u + 0x7FFFu + ((u >> 16) & 1u);
    return (unsigned short)(r >> 16);
}
__device__ __forceinline__ float bf16_to_f32(unsigned short h) {
    return __uint_as_float(((unsigned int)h) << 16);
}

// ---------------- fp32 -> bf16 weight conversion ----------------
__global__ __launch_bounds__(256) void cvt_bf16_kernel(const float* __restrict__ in,
                                                       unsigned short* __restrict__ out, int n) {
    int i = blockIdx.x * 256 + threadIdx.x;
    if (i < n) out[i] = f32_to_bf16(in[i]);
}

// ---------------- channelwise LayerNorm over C ----------------
__global__ __launch_bounds__(256) void ln_kernel(const float* __restrict__ x,
                                                 const float* __restrict__ gamma,
                                                 const float* __restrict__ beta,
                                                 float* __restrict__ u) {
    int idx = blockIdx.x * 256 + threadIdx.x;   // over B*T
    int b = idx >> 12;                          // T = 4096
    int t = idx & (TT - 1);
    const float* xp = x + (size_t)b * CC * TT + t;
    float s = 0.f, s2 = 0.f;
    #pragma unroll 4
    for (int c = 0; c < CC; ++c) {
        float v = xp[(size_t)c * TT];
        s += v; s2 = fmaf(v, v, s2);
    }
    float mean = s * (1.0f / CC);
    float var  = s2 * (1.0f / CC) - mean * mean;
    float rstd = rsqrtf(var + 1e-5f);
    float* up = u + (size_t)b * CC * TT + t;
    #pragma unroll 4
    for (int c = 0; c < CC; ++c) {
        float v = xp[(size_t)c * TT];
        up[(size_t)c * TT] = fmaf(gamma[c] * rstd, v - mean, beta[c]);
    }
}

// ---------------- S4D SSM scan + skip + exact GELU -> bf16 y ----------------
// One wave per (b,c). Lane j owns timesteps [128j, 128j+128); all 32 complex
// modes live in registers. Cross-lane Kogge-Stone scan stitches chunks.
__global__ __launch_bounds__(32) void ssm_kernel(const float* __restrict__ u,
                                                 const float* __restrict__ log_dt,
                                                 const float* __restrict__ C_re,
                                                 const float* __restrict__ C_im,
                                                 const float* __restrict__ log_A_real,
                                                 const float* __restrict__ A_imag,
                                                 const float* __restrict__ Dvec,
                                                 unsigned short* __restrict__ ybf) {
    __shared__ float lds[32 * 129];
    int bc = blockIdx.x;
    int c  = bc & (CC - 1);
    int lane = threadIdx.x;
    const float* urow = u + (size_t)bc * TT;

    // stage u into LDS (pitch 129 -> conflict-free chunk reads)
    for (int t = lane; t < TT; t += 32)
        lds[(t >> 7) * 129 + (t & 127)] = urow[t];
    __syncthreads();

    float dt = __expf(log_dt[c]);
    float lam_re[N2], lam_im[N2], cd_re[N2], cd_im[N2], P_re[N2], P_im[N2];
    #pragma unroll
    for (int n = 0; n < N2; ++n) {
        float Are = -__expf(log_A_real[c * N2 + n]);
        float Aim = A_imag[c * N2 + n];
        float dre = Are * dt, dim = Aim * dt;
        float m  = __expf(dre);
        float lr = m * __cosf(dim), li = m * __sinf(dim);
        lam_re[n] = lr; lam_im[n] = li;
        // Cd = (C_re + i C_im) * (lambda - 1) / A, folded with the factor 2
        float inv = 1.0f / (Are * Are + Aim * Aim);
        float nr = lr - 1.0f, ni = li;
        float tr = (nr * Are + ni * Aim) * inv;
        float ti = (ni * Are - nr * Aim) * inv;
        float cr = C_re[c * N2 + n], ci = C_im[c * N2 + n];
        cd_re[n] = 2.0f * (cr * tr - ci * ti);
        cd_im[n] = 2.0f * (cr * ti + ci * tr);
        // lambda^128 (chunk-combine factor)
        float m128 = __expf(128.0f * dre);
        float a128 = 128.0f * dim;
        P_re[n] = m128 * __cosf(a128);
        P_im[n] = m128 * __sinf(a128);
    }

    // Phase A: per-chunk end-state with zero init
    float s_re[N2], s_im[N2];
    #pragma unroll
    for (int n = 0; n < N2; ++n) { s_re[n] = 0.f; s_im[n] = 0.f; }
    const float* uc = &lds[lane * 129];
    #pragma unroll 1
    for (int i = 0; i < CHUNK; ++i) {
        float uv = uc[i];
        #pragma unroll
        for (int n = 0; n < N2; ++n) {
            float sr = s_re[n], si = s_im[n];
            float nsr = fmaf(lam_re[n], sr, fmaf(-lam_im[n], si, uv));
            float nsi = fmaf(lam_re[n], si, lam_im[n] * sr);
            s_re[n] = nsr; s_im[n] = nsi;
        }
    }

    // Kogge-Stone inclusive scan over lanes: s_j += P^(d) * s_{j-d}
    #pragma unroll
    for (int d = 1; d < 32; d <<= 1) {
        #pragma unroll
        for (int n = 0; n < N2; ++n) {
            float fr = __shfl_up(s_re[n], (unsigned)d, 32);
            float fi = __shfl_up(s_im[n], (unsigned)d, 32);
            if (lane >= d) {
                s_re[n] = fmaf(P_re[n], fr, fmaf(-P_im[n], fi, s_re[n]));
                s_im[n] = fmaf(P_re[n], fi, fmaf( P_im[n], fr, s_im[n]));
            }
        }
        #pragma unroll
        for (int n = 0; n < N2; ++n) {   // P *= P
            float pr = P_re[n], pi = P_im[n];
            P_re[n] = pr * pr - pi * pi;
            P_im[n] = 2.0f * pr * pi;
        }
    }
    // exclusive: chunk-start state = scan of lane-1 (lane 0 -> zero)
    #pragma unroll
    for (int n = 0; n < N2; ++n) {
        float pr = __shfl_up(s_re[n], 1u, 32);
        float pi = __shfl_up(s_im[n], 1u, 32);
        s_re[n] = (lane >= 1) ? pr : 0.0f;
        s_im[n] = (lane >= 1) ? pi : 0.0f;
    }

    // Phase C: replay chunk, emit y = gelu(2Re(Cd*s) + D*u), in-place in LDS
    float Dc = Dvec[c];
    float* yc = &lds[lane * 129];
    #pragma unroll 1
    for (int i = 0; i < CHUNK; ++i) {
        float uv = uc[i];
        float acc = 0.f;
        #pragma unroll
        for (int n = 0; n < N2; ++n) {
            float sr = s_re[n], si = s_im[n];
            float nsr = fmaf(lam_re[n], sr, fmaf(-lam_im[n], si, uv));
            float nsi = fmaf(lam_re[n], si, lam_im[n] * sr);
            s_re[n] = nsr; s_im[n] = nsi;
            acc = fmaf(cd_re[n], nsr, acc);
            acc = fmaf(-cd_im[n], nsi, acc);
        }
        float yv = fmaf(Dc, uv, acc);
        yv = 0.5f * yv * (1.0f + erff(yv * 0.70710678118654752f));  // exact GELU
        yc[i] = yv;
    }
    __syncthreads();
    unsigned short* yrow = ybf + (size_t)bc * TT;
    for (int t = lane; t < TT; t += 32)
        yrow[t] = f32_to_bf16(lds[(t >> 7) * 129 + (t & 127)]);
}

// ---------------- bf16 WMMA GEMM: out = epi(A[M,K] @ Bm[b][K,N] + bias) ----------------
// A tile staged with CDNA5 global_load_async_to_lds_b128 (ASYNCcnt path,
// no VGPR round trip). B tile (needs 16-bit transpose) staged through
// registers: loads issued before compute, LDS commit after compute.
// EPI 0: bf16 store (z for GLU). EPI 1: PReLU -> bf16. EPI 2: +residual -> f32.
#define APITCH 40
#define BPITCH 42

template <int EPI>
__global__ __launch_bounds__(256) void gemm_bf16_kernel(const unsigned short* __restrict__ A,
                                                        const unsigned short* __restrict__ Bm,
                                                        const float* __restrict__ bias,
                                                        const float* __restrict__ pa,
                                                        const float* __restrict__ resid,
                                                        void* __restrict__ outp,
                                                        int M, int N, int K) {
    __shared__ __align__(16) unsigned short Asm[2][128 * APITCH];
    __shared__ __align__(16) unsigned short Bsm[2][128 * BPITCH];

    int tid  = threadIdx.x;
    int lane = tid & 31, wave = tid >> 5;
    int hlf  = lane >> 4, l16 = lane & 15;
    int waveM = wave >> 1;       // 0..3 -> +32*waveM rows
    int waveN = wave & 1;        // 0..1 -> +64*waveN cols
    int n0 = blockIdx.x * 128, m0 = blockIdx.y * 128, b = blockIdx.z;
    const unsigned short* Bp = Bm + (size_t)b * K * N;

    // A staging: per-thread source row/quad, LDS dest offset (16B aligned)
    int aRow = tid >> 2, aQ = tid & 3;
    unsigned asmBase = (unsigned)(uintptr_t)&Asm[0][0];     // raw LDS offset (low 32 bits)
    auto stageA_async = [&](int buf, int kt) {
        #pragma unroll
        for (int i = 0; i < 2; ++i) {
            int row = aRow + 64 * i;
            unsigned ldsa = asmBase +
                (unsigned)((buf * 128 * APITCH + row * APITCH + aQ * 8) * 2);
            unsigned long long ga = (unsigned long long)(uintptr_t)
                (A + (size_t)(m0 + row) * K + kt * 32 + aQ * 8);
            asm volatile("global_load_async_to_lds_b128 %0, %1, off"
                         :: "v"(ldsa), "v"(ga) : "memory");
        }
    };

    // B staging through registers (transpose at LDS commit)
    unsigned int breg[8];
    auto loadB = [&](int kt) {
        #pragma unroll
        for (int i = 0; i < 8; ++i) {
            int d = tid + 256 * i;                    // 2048 dwords
            int k = d >> 6;
            int n = (d & 63) << 1;
            breg[i] = *(const unsigned int*)(Bp + (size_t)(kt * 32 + k) * N + n0 + n);
        }
    };
    auto storeB = [&](int buf) {
        #pragma unroll
        for (int i = 0; i < 8; ++i) {
            int d = tid + 256 * i;
            int k = d >> 6;
            int n = (d & 63) << 1;
            unsigned short* p = &Bsm[buf][n * BPITCH + k];
            p[0]      = (unsigned short)(breg[i] & 0xFFFFu);
            p[BPITCH] = (unsigned short)(breg[i] >> 16);
        }
    };

    v8f acc[2][4];
    #pragma unroll
    for (int mi = 0; mi < 2; ++mi)
        #pragma unroll
        for (int ni = 0; ni < 4; ++ni)
            acc[mi][ni] = (v8f){0.f, 0.f, 0.f, 0.f, 0.f, 0.f, 0.f, 0.f};

    // prologue: stage kt=0 into buffer 0
    loadB(0);
    stageA_async(0, 0);
    storeB(0);
    asm volatile("s_wait_asynccnt 0x0" ::: "memory");
    __syncthreads();

    int KT = K >> 5;
    for (int kt = 0; kt < KT; ++kt) {
        int buf = kt & 1;
        bool pf = (kt + 1 < KT);
        if (pf) {
            loadB(kt + 1);               // global loads in flight over compute
            stageA_async(buf ^ 1, kt + 1);  // async LDS writes in flight over compute
        }

        Frag afrag[2], bfrag[4];
        #pragma unroll
        for (int mi = 0; mi < 2; ++mi) {
            int m = 32 * waveM + 16 * mi + l16;
            const unsigned short* ap = &Asm[buf][m * APITCH];
            #pragma unroll
            for (int v = 0; v < 8; ++v) {             // k = 16*hlf + 2*v
                int k = (hlf << 4) + (v << 1);
                afrag[mi].u[v] = *(const unsigned int*)(ap + k);
            }
        }
        #pragma unroll
        for (int ni = 0; ni < 4; ++ni) {
            int n = 64 * waveN + 16 * ni + l16;
            const unsigned short* bp = &Bsm[buf][n * BPITCH + (hlf << 4)];
            #pragma unroll
            for (int v = 0; v < 8; ++v)               // k = 16*hlf + 2*v
                bfrag[ni].u[v] = *(const unsigned int*)(bp + 2 * v);
        }
        #pragma unroll
        for (int mi = 0; mi < 2; ++mi)
            #pragma unroll
            for (int ni = 0; ni < 4; ++ni)
                acc[mi][ni] = __builtin_amdgcn_wmma_f32_16x16x32_bf16(
                    false, afrag[mi].v, false, bfrag[ni].v,
                    (short)0, acc[mi][ni], false, false);

        if (pf) storeB(buf ^ 1);         // single batched loadcnt wait, behind WMMAs
        asm volatile("s_wait_asynccnt 0x0" ::: "memory");
        __syncthreads();
    }

    // epilogue: C/D layout -> row = base + r + 8*hlf, col = base + l16
    size_t ob = (size_t)b * M * N;
    #pragma unroll
    for (int mi = 0; mi < 2; ++mi) {
        #pragma unroll
        for (int ni = 0; ni < 4; ++ni) {
            #pragma unroll
            for (int r = 0; r < 8; ++r) {
                int row = m0 + 32 * waveM + 16 * mi + r + 8 * hlf;
                int col = n0 + 64 * waveN + 16 * ni + l16;
                float val = acc[mi][ni][r] + bias[row];
                size_t oidx = ob + (size_t)row * N + col;
                if (EPI == 1) {
                    val = (val >= 0.f) ? val : pa[row] * val;
                    ((unsigned short*)outp)[oidx] = f32_to_bf16(val);
                } else if (EPI == 2) {
                    ((float*)outp)[oidx] = val + resid[oidx];
                } else {
                    ((unsigned short*)outp)[oidx] = f32_to_bf16(val);
                }
            }
        }
    }
}

// ---------------- GLU over channel halves: v = z[:C] * sigmoid(z[C:]) ----------------
__global__ __launch_bounds__(256) void glu_kernel(const unsigned short* __restrict__ z,
                                                  unsigned short* __restrict__ v) {
    size_t idx = (size_t)blockIdx.x * 256 + threadIdx.x;      // over B*C*T
    size_t b   = idx >> 21;                                   // C*T = 2^21
    size_t rem = idx & ((1u << 21) - 1);
    const unsigned short* zb = z + (b << 22);                 // 2C*T = 2^22
    float z1 = bf16_to_f32(zb[rem]);
    float z2 = bf16_to_f32(zb[rem + (1u << 21)]);
    float s  = 1.0f / (1.0f + __expf(-z2));
    v[idx] = f32_to_bf16(z1 * s);
}

extern "C" void kernel_launch(void* const* d_in, const int* in_sizes, int n_in,
                              void* d_out, int out_size, void* d_ws, size_t ws_size,
                              hipStream_t stream) {
    const float* x          = (const float*)d_in[0];
    const float* gamma      = (const float*)d_in[1];
    const float* beta       = (const float*)d_in[2];
    const float* log_dt     = (const float*)d_in[3];
    const float* C_re       = (const float*)d_in[4];
    const float* C_im       = (const float*)d_in[5];
    const float* log_A_real = (const float*)d_in[6];
    const float* A_imag     = (const float*)d_in[7];
    const float* Dv         = (const float*)d_in[8];
    const float* Wo         = (const float*)d_in[9];
    const float* bo         = (const float*)d_in[10];
    const float* W1         = (const float*)d_in[11];
    const float* b1         = (const float*)d_in[12];
    const float* prelu_a    = (const float*)d_in[13];
    const float* W2         = (const float*)d_in[14];
    const float* b2         = (const float*)d_in[15];

    char* ws = (char*)d_ws;
    size_t off = 0;
    float* u = (float*)(ws + off);                   // 33.5 MB; dead after ssm
    unsigned short* zbuf = (unsigned short*)u;       // alias: z bf16 [B,2C,T] = 33.5 MB
    off += (size_t)BB * CC * TT * sizeof(float);
    unsigned short* ybuf = (unsigned short*)(ws + off);  // y bf16; dead after gemm0
    unsigned short* vbuf = ybuf;                         // alias: GLU output
    off += (size_t)BB * CC * TT * sizeof(unsigned short);
    unsigned short* hbuf = (unsigned short*)(ws + off);  // h bf16 [B,H,T] = 67 MB
    off += (size_t)BB * HH * TT * sizeof(unsigned short);
    unsigned short* Wo_b = (unsigned short*)(ws + off); off += (size_t)2 * CC * CC * 2;
    unsigned short* W1_b = (unsigned short*)(ws + off); off += (size_t)HH * CC * 2;
    unsigned short* W2_b = (unsigned short*)(ws + off); off += (size_t)CC * HH * 2;

    // weights -> bf16
    cvt_bf16_kernel<<<(2 * CC * CC + 255) / 256, 256, 0, stream>>>(Wo, Wo_b, 2 * CC * CC);
    cvt_bf16_kernel<<<(HH * CC + 255) / 256, 256, 0, stream>>>(W1, W1_b, HH * CC);
    cvt_bf16_kernel<<<(CC * HH + 255) / 256, 256, 0, stream>>>(W2, W2_b, CC * HH);

    // LayerNorm over C
    ln_kernel<<<(BB * TT) / 256, 256, 0, stream>>>(x, gamma, beta, u);

    // S4D SSM (chunked scan) + D-skip + exact GELU -> y bf16
    ssm_kernel<<<BB * CC, 32, 0, stream>>>(u, log_dt, C_re, C_im, log_A_real,
                                           A_imag, Dv, ybuf);

    // z = Wo @ y + bo   (M=1024, K=512, N=4096, batched over B)
    gemm_bf16_kernel<0><<<dim3(TT / 128, (2 * CC) / 128, BB), 256, 0, stream>>>(
        Wo_b, ybuf, bo, bo, x, (void*)zbuf, 2 * CC, TT, CC);

    // GLU
    glu_kernel<<<((size_t)BB * CC * TT) / 256, 256, 0, stream>>>(zbuf, vbuf);

    // h = prelu(W1 @ v + b1)   (M=2048, K=512)
    gemm_bf16_kernel<1><<<dim3(TT / 128, HH / 128, BB), 256, 0, stream>>>(
        W1_b, vbuf, b1, prelu_a, x, (void*)hbuf, HH, TT, CC);

    // out = x + W2 @ h + b2    (M=512, K=2048), f32 output with residual
    gemm_bf16_kernel<2><<<dim3(TT / 128, CC / 128, BB), 256, 0, stream>>>(
        W2_b, hbuf, b2, b2, x, d_out, CC, TT, HH);
}
